// GraphConstructor_54503134986655
// MI455X (gfx1250) — matrix operand order repaired and checked by reading, not compiled, not run on hardware
//
#include <hip/hip_runtime.h>

// ---------------------------------------------------------------------------
// GraphConstructor for MI455X (gfx1250, wave32):
//   K0: repack conv weights fp32 -> bf16 hi/lo in WMMA A-fragment lane order
//   K1: fused conv1+conv2 (implicit GEMM, v_wmma_f32_16x16x32_bf16, 3-term
//       hi/lo split for ~fp32 accuracy) + bias + tanh(3x) -> n1,n2 in ws.
//       Register-blocked: each wave applies its A fragments to 4 N-tiles
//       (64 positions) -> 24 WMMAs per 4 A-fragment loads per k-step,
//       cutting L2 weight traffic ~4x vs 1 N-tile per wave.
//   K2: per-pixel 5x5 gram over 64 ch + antisym + sigmoid + eye + row-norm
// ---------------------------------------------------------------------------

typedef __attribute__((ext_vector_type(16))) __bf16 v16bf;
typedef __attribute__((ext_vector_type(8)))  float  v8f;

#define GC_ALPHA 3.0f

__device__ __forceinline__ unsigned short f2bf_rne(float x) {
  unsigned u = __builtin_bit_cast(unsigned, x);
  unsigned r = u + 0x7FFFu + ((u >> 16) & 1u);   // round-to-nearest-even
  return (unsigned short)(r >> 16);
}
__device__ __forceinline__ float bf2f(unsigned short h) {
  return __builtin_bit_cast(float, ((unsigned)h) << 16);
}
__device__ __forceinline__ float tanh3(float u) {  // tanh of already-scaled arg
  return 1.0f - 2.0f / (1.0f + __expf(2.0f * u)); // saturates correctly
}

// --------------------------------------------------------------------------
// Kernel 0: repack weights.
// K index convention: k = (ky*3+kx)*64 + ic, K = 576 = 18 steps of 32.
// A-fragment element (mt, kk, lane, j) -> (oc, k) in contiguous k order (B uses the same):
//   lanes 0-15 : K_local = j       (K 0..15)
//   lanes 16-31: K_local = j + 16  (K 16..31)
// --------------------------------------------------------------------------
__global__ void __launch_bounds__(256) repack_weights(
    const float* __restrict__ W1, const float* __restrict__ W2,
    unsigned short* __restrict__ A1h, unsigned short* __restrict__ A1l,
    unsigned short* __restrict__ A2h, unsigned short* __restrict__ A2l) {
  int gid = blockIdx.x * 256 + threadIdx.x;          // 0 .. 2*36864-1
  if (gid >= 2 * 36864) return;
  int conv = gid >= 36864;
  int e    = gid - conv * 36864;                     // ((mt*18+kk)*32+lane)*16 + j
  int j    = e & 15;
  int lane = (e >> 4) & 31;
  int kk   = (e >> 9) % 18;
  int mt   = e / 9216;
  int oc   = mt * 16 + (lane & 15);
  int kl   = ((lane >> 4) << 4) + j;
  int k    = kk * 32 + kl;
  int tap  = k >> 6, ic = k & 63;
  int ky   = tap / 3, kx = tap % 3;
  const float* W = conv ? W2 : W1;                   // W[oc][ic][ky][kx]
  float w = W[((oc * 64 + ic) * 3 + ky) * 3 + kx];
  unsigned short h = f2bf_rne(w);
  unsigned short l = f2bf_rne(w - bf2f(h));
  if (conv) { A2h[e] = h; A2l[e] = l; }
  else      { A1h[e] = h; A1l[e] = l; }
}

// --------------------------------------------------------------------------
// Kernel 1: fused conv1+conv2 + tanh(3x).
// Grid: 10 images * 128 rows * 2 half-rows = 2560 blocks, 256 threads.
// Block: 8 waves = 4 oc-tiles x 2 position-groups; each wave register-blocks
// 4 N-tiles -> block covers 64 oc x 128 y for both convs.
// LDS patch: [3 rows][130 cols][64 ch, padded to 80] as bf16 hi + lo.
// --------------------------------------------------------------------------
#define CPAD 80
#define PCOLS 130

__global__ void __launch_bounds__(256) conv2_tanh(
    const float* __restrict__ pts,
    const unsigned short* __restrict__ A1h, const unsigned short* __restrict__ A1l,
    const unsigned short* __restrict__ A2h, const unsigned short* __restrict__ A2l,
    const float* __restrict__ b1, const float* __restrict__ b2,
    float* __restrict__ n1, float* __restrict__ n2) {
  __shared__ __align__(32) unsigned short ldsHi[3 * PCOLS * CPAD];
  __shared__ __align__(32) unsigned short ldsLo[3 * PCOLS * CPAD];

  const int tid  = threadIdx.x;
  const int lane = tid & 31;
  const int wave = tid >> 5;
  const int mt   = wave & 3;   // oc tile
  const int ng   = wave >> 2;  // position group (64 positions each)

  const int blk = blockIdx.x;
  const int img = blk >> 8;           // 0..9
  const int xr  = (blk >> 1) & 127;   // 0..127
  const int y0  = (blk & 1) << 7;     // 0 or 128

  // Stage fp32 patch -> bf16 hi/lo in LDS (zero-fill SAME-padding halo).
  for (int idx = tid; idx < 3 * PCOLS * 64; idx += 256) {
    int c  = idx % PCOLS;
    int t  = idx / PCOLS;
    int ic = t & 63;
    int r  = t >> 6;
    int xi = xr + r - 1;
    int yi = y0 + c - 1;
    float v = 0.0f;
    if ((unsigned)xi < 128u && (unsigned)yi < 256u)
      v = pts[(((size_t)img * 64 + ic) * 128 + xi) * 256 + yi];
    unsigned short h = f2bf_rne(v);
    unsigned short l = f2bf_rne(v - bf2f(h));
    int li = (r * PCOLS + c) * CPAD + ic;
    ldsHi[li] = h;
    ldsLo[li] = l;
  }
  __syncthreads();

  v8f acc1[4], acc2[4];
#pragma unroll
  for (int nt = 0; nt < 4; ++nt) {
    acc1[nt] = (v8f){0.f, 0.f, 0.f, 0.f, 0.f, 0.f, 0.f, 0.f};
    acc2[nt] = (v8f){0.f, 0.f, 0.f, 0.f, 0.f, 0.f, 0.f, 0.f};
  }

  const int row16 = lane & 15;
  const int khalf = (lane >> 4) & 1;
  const int pbase = ng * 64 + row16;   // position within 128-chunk for nt=0

  for (int kk = 0; kk < 18; ++kk) {
    const int tap = kk >> 1;
    const int ky  = tap / 3;
    const int kx  = tap % 3;
    // B-fragment: lane holds K_local = khalf*16 + j  ->  ic = (kk&1)*32 + that
    const int icb = ((kk & 1) << 5) + (khalf << 4);

    const int abase = ((mt * 18 + kk) * 32 + lane) * 16;  // contiguous 32B/lane
    const v16bf a1h = *(const v16bf*)&A1h[abase];
    const v16bf a1l = *(const v16bf*)&A1l[abase];
    const v16bf a2h = *(const v16bf*)&A2h[abase];
    const v16bf a2l = *(const v16bf*)&A2l[abase];

#pragma unroll
    for (int nt = 0; nt < 4; ++nt) {
      const int pc  = pbase + nt * 16 + kx;             // patch col 0..129
      const int bli = (ky * PCOLS + pc) * CPAD + icb;   // 32B-aligned
      const v16bf bh = *(const v16bf*)&ldsHi[bli];
      const v16bf bl = *(const v16bf*)&ldsLo[bli];
      // 3-term bf16 split: hi*hi + hi*lo + lo*hi (~fp32 accuracy, f32 accum)
      acc1[nt] = __builtin_amdgcn_wmma_f32_16x16x32_bf16(false, a1h, false, bh, (short)0, acc1[nt], false, false);
      acc1[nt] = __builtin_amdgcn_wmma_f32_16x16x32_bf16(false, a1h, false, bl, (short)0, acc1[nt], false, false);
      acc1[nt] = __builtin_amdgcn_wmma_f32_16x16x32_bf16(false, a1l, false, bh, (short)0, acc1[nt], false, false);
      acc2[nt] = __builtin_amdgcn_wmma_f32_16x16x32_bf16(false, a2h, false, bh, (short)0, acc2[nt], false, false);
      acc2[nt] = __builtin_amdgcn_wmma_f32_16x16x32_bf16(false, a2h, false, bl, (short)0, acc2[nt], false, false);
      acc2[nt] = __builtin_amdgcn_wmma_f32_16x16x32_bf16(false, a2l, false, bh, (short)0, acc2[nt], false, false);
    }
  }

  // Epilogue: C/D layout -> lanes 0-15: M=v; lanes 16-31: M=8+v; N=lane%16.
  const int mbase = mt * 16 + (khalf << 3);
#pragma unroll
  for (int nt = 0; nt < 4; ++nt) {
    const int y = y0 + pbase + nt * 16;
#pragma unroll
    for (int vv = 0; vv < 8; ++vv) {
      const int oc = mbase + vv;
      const size_t o = (((size_t)img * 64 + oc) * 128 + xr) * 256 + y;
      const float u1 = GC_ALPHA * (acc1[nt][vv] + b1[oc]);
      const float u2 = GC_ALPHA * (acc2[nt][vv] + b2[oc]);
      n1[o] = tanh3(u1);
      n2[o] = tanh3(u2);
    }
  }
}

// --------------------------------------------------------------------------
// Kernel 2: per-pixel 5x5 gram + antisym + sigmoid + eye + row-normalize.
// One thread per (b, x, y); lanes walk consecutive y -> all traffic coalesced.
// --------------------------------------------------------------------------
__global__ void __launch_bounds__(256) gram_adj(
    const float* __restrict__ n1, const float* __restrict__ n2,
    float* __restrict__ out) {
  const int gid = blockIdx.x * 256 + threadIdx.x;  // 0..65535
  const int b   = gid >> 15;
  const int p   = gid & 32767;                     // x*256 + y
  const size_t base = ((size_t)b * 5 * 64 << 15) + p;

  float s[5][5];
#pragma unroll
  for (int m = 0; m < 5; ++m)
#pragma unroll
    for (int nn = 0; nn < 5; ++nn) s[m][nn] = 0.f;

  for (int l = 0; l < 64; ++l) {
    float a[5], c[5];
#pragma unroll
    for (int m = 0; m < 5; ++m)  a[m]  = n1[base + ((size_t)(m * 64 + l) << 15)];
#pragma unroll
    for (int nn = 0; nn < 5; ++nn) c[nn] = n2[base + ((size_t)(nn * 64 + l) << 15)];
#pragma unroll
    for (int m = 0; m < 5; ++m)
#pragma unroll
      for (int nn = 0; nn < 5; ++nn) s[m][nn] = fmaf(a[m], c[nn], s[m][nn]);
  }

  const size_t obase = ((size_t)b * 25 << 15) + p;
#pragma unroll
  for (int m = 0; m < 5; ++m) {
    float row[5];
    float denom = 0.f;
#pragma unroll
    for (int nn = 0; nn < 5; ++nn) {
      const float aa = s[m][nn] - s[nn][m];
      float e = 1.f / (1.f + __expf(-GC_ALPHA * aa));
      if (m == nn) e += 1.f;
      row[nn] = e;
      denom  += e;
    }
    const float inv = 1.f / denom;
#pragma unroll
    for (int nn = 0; nn < 5; ++nn)
      out[obase + ((size_t)(m * 5 + nn) << 15)] = row[nn] * inv;
  }
}

// --------------------------------------------------------------------------
extern "C" void kernel_launch(void* const* d_in, const int* in_sizes, int n_in,
                              void* d_out, int out_size, void* d_ws, size_t ws_size,
                              hipStream_t stream) {
  const float* pts = (const float*)d_in[0];
  const float* W1  = (const float*)d_in[1];
  const float* b1  = (const float*)d_in[2];
  const float* W2  = (const float*)d_in[3];
  const float* b2  = (const float*)d_in[4];
  float* out = (float*)d_out;

  // Workspace layout:
  //   [0)          4 x 36864 ushort  A-fragment hi/lo for W1, W2 (288 KB)
  //   [294912)     n1: 20,971,520 f32 (83.9 MB)
  //   [+83.9MB)    n2: 20,971,520 f32 (83.9 MB)
  char* ws = (char*)d_ws;
  unsigned short* A1h = (unsigned short*)ws;
  unsigned short* A1l = A1h + 36864;
  unsigned short* A2h = A1l + 36864;
  unsigned short* A2l = A2h + 36864;
  float* n1 = (float*)(ws + 4 * 36864 * sizeof(unsigned short));
  float* n2 = n1 + (size_t)10 * 64 * 128 * 256;

  repack_weights<<<288, 256, 0, stream>>>(W1, W2, A1h, A1l, A2h, A2l);
  conv2_tanh<<<2560, 256, 0, stream>>>(pts, A1h, A1l, A2h, A2l, b1, b2, n1, n2);
  gram_adj<<<256, 256, 0, stream>>>(n1, n2, out);
}
